// MultiheadRCDA_71485435675153
// MI455X (gfx1250) — compile-verified, matrix operand order checked by reading, not run on hardware
//
#include <hip/hip_runtime.h>
#include <stdint.h>

// ---------------------------------------------------------------- types
typedef __attribute__((ext_vector_type(8)))  float          v8f;
typedef __attribute__((ext_vector_type(16))) __bf16         v16bf;
typedef __attribute__((ext_vector_type(4)))  unsigned int   u32x4;
typedef __attribute__((ext_vector_type(8)))  int            i32x8;
typedef __attribute__((ext_vector_type(4)))  int            i32x4;

union ABFrag { unsigned short h[16]; v16bf v; };

#ifndef __has_builtin
#define __has_builtin(x) 0
#endif
#if __has_builtin(__builtin_amdgcn_tensor_load_to_lds)
#define USE_TDM 1
#else
#define USE_TDM 0
#endif

// Native bf16 converts (v_cvt_bf16_f32 / packed) instead of integer emulation.
__device__ __forceinline__ unsigned short f2bf(float f) {
  __bf16 h = (__bf16)f;
  return __builtin_bit_cast(unsigned short, h);
}
__device__ __forceinline__ float bf2f(unsigned short u) {
  __bf16 h = __builtin_bit_cast(__bf16, u);
  return (float)h;
}

// N=8 batch, L=300, H=W=96, E=256, 8 heads, head_dim=32.  L padded to 304.

// ---------------------------------------------------------------- kernel A
__global__ __launch_bounds__(256) void mean_hw_kernel(
    const float* __restrict__ kr, const float* __restrict__ kc,
    float* __restrict__ krm, float* __restrict__ kcm) {
  int bi = blockIdx.x;            // 0..767
  int b = bi / 96, j = bi % 96;
  int e = threadIdx.x;            // 0..255
  float s1 = 0.f, s2 = 0.f;
  for (int t = 0; t < 96; ++t) {
    s1 += kr[(((size_t)b * 96 + t) * 96 + j) * 256 + e];   // mean over h (axis 1)
    s2 += kc[(((size_t)b * 96 + j) * 96 + t) * 256 + e];   // mean over w (axis 2)
  }
  krm[((size_t)b * 96 + j) * 256 + e] = s1 * (1.f / 96.f);
  kcm[((size_t)b * 96 + j) * 256 + e] = s2 * (1.f / 96.f);
}

// ---------------------------------------------------------------- kernel B
// Y = (X[M,256] @ W[256,256]^T + bias) * scale ; bf16 WMMA, f32 accum.
// Each wave computes a 16x64 strip: one A-frag feeds 4 WMMAs per k-step.
// OUTMODE 0: bf16 row-major (ld 256).  OUTMODE 1: f32, row (b*Ldim+l) -> (l*8+b).
template <int OUTMODE>
__global__ __launch_bounds__(256) void wmma_gemm256(
    const float* __restrict__ X, const float* __restrict__ W,
    const float* __restrict__ Bv, float scale, void* __restrict__ Y,
    int M, int Ldim) {
  int wid  = threadIdx.x >> 5;
  int lane = threadIdx.x & 31;
  long task = (long)blockIdx.x * 8 + wid;        // total tasks = M/4, grid = M/32
  int mtile  = (int)(task >> 2);
  int ngroup = (int)(task & 3);                  // 64-wide N strip
  if (mtile * 16 >= M) return;

  const int mloc = lane & 15;
  const int kb   = (lane < 16) ? 0 : 8;          // 16-bit A/B frag K-split
  const int m    = mtile * 16 + mloc;
  const float* ap = X + (size_t)m * 256 + kb;
  const float* bp[4];
#pragma unroll
  for (int j = 0; j < 4; ++j)
    bp[j] = W + (size_t)(ngroup * 64 + j * 16 + mloc) * 256 + kb;  // B[k][n]=W[n][k]

  v8f acc[4] = {};
#pragma unroll
  for (int k0 = 0; k0 < 256; k0 += 32) {
    ABFrag a;
#pragma unroll
    for (int i = 0; i < 8; ++i) {
      a.h[i]     = f2bf(ap[k0 + i]);
      a.h[8 + i] = f2bf(ap[k0 + 16 + i]);
    }
#pragma unroll
    for (int j = 0; j < 4; ++j) {
      ABFrag b;
#pragma unroll
      for (int i = 0; i < 8; ++i) {
        b.h[i]     = f2bf(bp[j][k0 + i]);
        b.h[8 + i] = f2bf(bp[j][k0 + 16 + i]);
      }
      acc[j] = __builtin_amdgcn_wmma_f32_16x16x32_bf16(false, a.v, false, b.v,
                                                       (short)0, acc[j], false, false);
    }
  }
#pragma unroll
  for (int j = 0; j < 4; ++j) {
    int n = ngroup * 64 + j * 16 + mloc;
    float bias = Bv[n];
#pragma unroll
    for (int r = 0; r < 8; ++r) {
      int row = mtile * 16 + r + ((lane < 16) ? 0 : 8);
      float val = (acc[j][r] + bias) * scale;
      if (OUTMODE == 0) {
        ((unsigned short*)Y)[(size_t)row * 256 + n] = f2bf(val);
      } else {
        int bb = row / Ldim, l = row % Ldim;
        ((float*)Y)[((size_t)l * 8 + bb) * 256 + n] = val;
      }
    }
  }
}

// ---------------------------------------------------------------- kernel C
// scores[l,w] = sum_d q[b,l,n,d]*k[b,w,n,d], softmax over w, store bf16.
__global__ __launch_bounds__(32) void scores_softmax_kernel(
    const unsigned short* __restrict__ qp, const unsigned short* __restrict__ kp,
    unsigned short* __restrict__ wout) {
  int lt = blockIdx.x % 19;
  int bn = blockIdx.x / 19;
  int b = bn >> 3, n = bn & 7;
  int lane = threadIdx.x & 31;
  int mloc = lane & 15;
  int kb   = (lane < 16) ? 0 : 8;
  int m0   = lt * 16;

  __shared__ float s[16][96];

  ABFrag a;
  bool arow_ok = (m0 + mloc) < 300;
  const unsigned short* apq =
      qp + ((size_t)(b * 300 + (arow_ok ? (m0 + mloc) : 0))) * 256 + n * 32 + kb;
#pragma unroll
  for (int i = 0; i < 8; ++i) {
    a.h[i]     = arow_ok ? apq[i] : (unsigned short)0;
    a.h[8 + i] = arow_ok ? apq[16 + i] : (unsigned short)0;
  }
#pragma unroll
  for (int j = 0; j < 6; ++j) {
    const unsigned short* bp =
        kp + ((size_t)(b * 96 + j * 16 + mloc)) * 256 + n * 32 + kb;
    ABFrag bb;
#pragma unroll
    for (int i = 0; i < 8; ++i) { bb.h[i] = bp[i]; bb.h[8 + i] = bp[16 + i]; }
    v8f c = {};
    c = __builtin_amdgcn_wmma_f32_16x16x32_bf16(false, a.v, false, bb.v,
                                                (short)0, c, false, false);
#pragma unroll
    for (int r = 0; r < 8; ++r) {
      int rr = r + ((lane < 16) ? 0 : 8);
      s[rr][j * 16 + mloc] = c[r];
    }
  }
  __syncthreads();
  if (lane < 16) {
    int l = m0 + lane;
    unsigned short* op = wout + ((size_t)bn * 304 + l) * 96;
    if (l < 300) {
      float mx = -1e30f;
      for (int w = 0; w < 96; ++w) mx = fmaxf(mx, s[lane][w]);
      float sum = 0.f;
      for (int w = 0; w < 96; ++w) { float e = __expf(s[lane][w] - mx); s[lane][w] = e; sum += e; }
      float inv = 1.f / sum;
      for (int w = 0; w < 96; ++w) op[w] = f2bf(s[lane][w] * inv);
    } else {
      for (int w = 0; w < 96; ++w) op[w] = 0;
    }
  }
}

// ---------------------------------------------------------------- kernel D
// out[b,l,n*32+d] = sum_{h,w} wcol[l,h]*wrow[l,w]*vp[b,h,w,n*32+d]
// as GEMM: A[l, h*96+w] = wcol*wrow (built on the fly), B = vp slice (9216x32).
// B tiles stream through double-buffered LDS via the Tensor Data Mover.
__global__ __launch_bounds__(256) void rcda_attn_kernel(
    const unsigned short* __restrict__ vp, const unsigned short* __restrict__ wrow,
    const unsigned short* __restrict__ wcol, float* __restrict__ attn) {
  int bn = blockIdx.x;
  int b = bn >> 3, n = bn & 7;
  int wid  = threadIdx.x >> 5;
  int lane = threadIdx.x & 31;
  int mt_raw = blockIdx.y * 8 + wid;
  int mtile  = (mt_raw < 19) ? mt_raw : 18;
  bool ok    = mt_raw < 19;
  int m0 = mtile * 16;
  int mloc = lane & 15;
  int kb   = (lane < 16) ? 0 : 8;

  __shared__ unsigned short lrow[8][16][96];
  __shared__ unsigned short lcol[8][16][96];
  __shared__ unsigned short ldsB[2][32][32];   // double-buffered 2KB B tiles

  {  // stage this wave's softmax tiles (16 rows x 96) into LDS
    int r = lane & 15, half = lane >> 4;
    const unsigned short* sr = wrow + ((size_t)bn * 304 + m0 + r) * 96 + half * 48;
    const unsigned short* sc = wcol + ((size_t)bn * 304 + m0 + r) * 96 + half * 48;
    for (int q = 0; q < 48; ++q) {
      lrow[wid][r][half * 48 + q] = sr[q];
      lcol[wid][r][half * 48 + q] = sc[q];
    }
  }

  size_t vbase = ((size_t)b * 9216) * 256 + (size_t)n * 32;     // elements

#if USE_TDM
  unsigned lds_base = (unsigned)(unsigned long long)(void*)&ldsB[0][0][0];
  auto tdm_issue = [&](int t, int buf) {
    unsigned long long ga =
        (unsigned long long)(const void*)vp + (vbase + (size_t)t * 32 * 256) * 2ull;
    u32x4 g0 = {};
    g0[0] = 1u;                                   // count = 1 descriptor
    g0[1] = lds_base + (unsigned)buf * 2048u;     // LDS byte address
    g0[2] = (unsigned)ga;                         // global_addr[31:0]
    g0[3] = (unsigned)((ga >> 32) & 0x1FFFFFFull) | 0x80000000u;  // type=2
    i32x8 g1 = {};
    g1[0] = 1 << 16;      // data_size = 2 bytes
    g1[1] = 32 << 16;     // tensor_dim0 = 32
    g1[2] = 32 << 16;     // tensor_dim1 = 32
    g1[3] = 32 << 16;     // tile_dim0 = 32
    g1[4] = 32;           // tile_dim1 = 32, tile_dim2 = 0
    g1[5] = 256;          // tensor_dim0_stride = 256 elements
    g1[6] = 0;
    g1[7] = 0;
    i32x4 z4 = {};
#if defined(__clang_major__) && (__clang_major__ >= 23)
    i32x8 z8 = {};
    __builtin_amdgcn_tensor_load_to_lds(g0, g1, z4, z4, z8, 0);
#else
    __builtin_amdgcn_tensor_load_to_lds(g0, g1, z4, z4, 0);
#endif
  };
  if (wid == 0) tdm_issue(0, 0);       // prologue: prefetch tile 0
#endif
  __syncthreads();

  v8f acc0 = {}, acc1 = {};

  for (int t = 0; t < 288; ++t) {          // K = 9216 = 288 * 32
    int buf = t & 1;
#if USE_TDM
    if (wid == 0) {
      if (t + 1 < 288) {
        tdm_issue(t + 1, (t + 1) & 1);                 // overlap DMA w/ compute
        __builtin_amdgcn_s_wait_tensorcnt((short)1);   // in-order: tile t landed
      } else {
        __builtin_amdgcn_s_wait_tensorcnt((short)0);
      }
    }
#else
    {  // cooperative B-tile load: 32 rows x 64B, one 8B chunk per thread
      int row = threadIdx.x >> 3;
      int off = (threadIdx.x & 7) * 4;
      const unsigned short* src = vp + vbase + ((size_t)t * 32 + row) * 256 + off;
      *(uint2*)&ldsB[buf][row][off] = *(const uint2*)src;
    }
#endif
    __syncthreads();

    int hh = t / 3;
    int w0 = (t - hh * 3) * 32;
    float cc = bf2f(lcol[wid][mloc][hh]);
    ABFrag a;
#pragma unroll
    for (int i = 0; i < 8; ++i) {
      a.h[i]     = f2bf(cc * bf2f(lrow[wid][mloc][w0 + kb + i]));
      a.h[8 + i] = f2bf(cc * bf2f(lrow[wid][mloc][w0 + kb + 16 + i]));
    }
    ABFrag b0, b1;
#pragma unroll
    for (int i = 0; i < 8; ++i) {
      b0.h[i]     = ldsB[buf][kb + i][mloc];
      b0.h[8 + i] = ldsB[buf][kb + 16 + i][mloc];
      b1.h[i]     = ldsB[buf][kb + i][16 + mloc];
      b1.h[8 + i] = ldsB[buf][kb + 16 + i][16 + mloc];
    }
    acc0 = __builtin_amdgcn_wmma_f32_16x16x32_bf16(false, a.v, false, b0.v,
                                                   (short)0, acc0, false, false);
    acc1 = __builtin_amdgcn_wmma_f32_16x16x32_bf16(false, a.v, false, b1.v,
                                                   (short)0, acc1, false, false);
    __syncthreads();
  }

  if (ok) {
#pragma unroll
    for (int r = 0; r < 8; ++r) {
      int row = m0 + r + ((lane < 16) ? 0 : 8);
      if (row < 300) {
        float* op = attn + ((size_t)(b * 300 + row)) * 256 + n * 32;
        op[mloc]      = acc0[r];
        op[16 + mloc] = acc1[r];
      }
    }
  }
}

// ---------------------------------------------------------------- launcher
extern "C" void kernel_launch(void* const* d_in, const int* in_sizes, int n_in,
                              void* d_out, int out_size, void* d_ws, size_t ws_size,
                              hipStream_t stream) {
  (void)in_sizes; (void)n_in; (void)out_size; (void)ws_size;
  const float* query_row = (const float*)d_in[0];   // (8,300,256)
  const float* query_col = (const float*)d_in[1];
  const float* key_row   = (const float*)d_in[2];   // (8,96,96,256)
  const float* key_col   = (const float*)d_in[3];
  const float* value     = (const float*)d_in[4];
  const float* ipw       = (const float*)d_in[5];   // (1280,256)
  const float* ipb       = (const float*)d_in[6];   // (1280,)
  const float* opw       = (const float*)d_in[7];   // (256,256)
  const float* opb       = (const float*)d_in[8];

  char* base = (char*)d_ws;
  size_t o = 0;
  auto take = [&](size_t bytes) { char* p = base + o; o = (o + bytes + 255) & ~(size_t)255; return p; };
  float*          krm  = (float*)take((size_t)768 * 256 * 4);
  float*          kcm  = (float*)take((size_t)768 * 256 * 4);
  unsigned short* qrp  = (unsigned short*)take((size_t)2400 * 256 * 2);
  unsigned short* qcp  = (unsigned short*)take((size_t)2400 * 256 * 2);
  unsigned short* krp  = (unsigned short*)take((size_t)768 * 256 * 2);
  unsigned short* kcp  = (unsigned short*)take((size_t)768 * 256 * 2);
  unsigned short* vp   = (unsigned short*)take((size_t)73728 * 256 * 2);
  unsigned short* wro  = (unsigned short*)take((size_t)64 * 304 * 96 * 2);
  unsigned short* wco  = (unsigned short*)take((size_t)64 * 304 * 96 * 2);
  float*          attn = (float*)take((size_t)2400 * 256 * 4);

  const float scaling = 0.17677669529663687f;   // 32^-0.5

  // 1) column/row means of keys (projection is linear, so mean-then-project)
  mean_hw_kernel<<<768, 256, 0, stream>>>(key_row, key_col, krm, kcm);

  // 2) projections (grid.x = M/32: each wave does 16x64, 8 waves per block)
  wmma_gemm256<0><<<75,   256, 0, stream>>>(query_row, ipw,             ipb,        scaling, qrp, 2400, 300);
  wmma_gemm256<0><<<75,   256, 0, stream>>>(query_col, ipw + 1 * 65536, ipb + 256,  scaling, qcp, 2400, 300);
  wmma_gemm256<0><<<24,   256, 0, stream>>>(krm,       ipw + 2 * 65536, ipb + 512,  1.0f,    krp, 768,  0);
  wmma_gemm256<0><<<24,   256, 0, stream>>>(kcm,       ipw + 3 * 65536, ipb + 768,  1.0f,    kcp, 768,  0);
  wmma_gemm256<0><<<2304, 256, 0, stream>>>(value,     ipw + 4 * 65536, ipb + 1024, 1.0f,    vp,  73728, 0);

  // 3) attention weights (row over w, col over h), softmaxed, bf16
  scores_softmax_kernel<<<64 * 19, 32, 0, stream>>>(qrp, krp, wro);
  scores_softmax_kernel<<<64 * 19, 32, 0, stream>>>(qcp, kcp, wco);

  // 4) fused bilinear contraction per (b, head), TDM-fed double-buffered LDS
  rcda_attn_kernel<<<dim3(64, 3), 256, 0, stream>>>(vp, wro, wco, attn);

  // 5) output projection with (b,l)->(l,b) permuted f32 store into d_out
  wmma_gemm256<1><<<75, 256, 0, stream>>>(attn, opw, opb, 1.0f, d_out, 2400, 300);
}
